// SelfAttention_21380347200129
// MI455X (gfx1250) — matrix operand order repaired and checked by reading, not compiled, not run on hardware
//
#include <hip/hip_runtime.h>

// ---------------------------------------------------------------- types
typedef __bf16 bf16;
typedef __attribute__((ext_vector_type(16))) __bf16 v16bf;
typedef __attribute__((ext_vector_type(8)))  float  v8f;
typedef __attribute__((ext_vector_type(4)))  unsigned int u32x4;

union Frag { v16bf v; u32x4 u[2]; };

__device__ __forceinline__ bf16 f2bf(float f) {
  unsigned u = __builtin_bit_cast(unsigned, f);
  u += 0x7FFFu + ((u >> 16) & 1u);           // round to nearest even
  unsigned short h = (unsigned short)(u >> 16);
  return __builtin_bit_cast(bf16, h);
}

// async global->LDS copy of 16 bytes, tracked by ASYNCcnt
__device__ __forceinline__ void async_b128(const void* gptr, void* lptr) {
  unsigned lds = (unsigned)(uintptr_t)lptr;   // low 32 bits = LDS byte address
  asm volatile("global_load_async_to_lds_b128 %0, %1, off"
               :: "v"(lds), "v"(gptr) : "memory");
}
__device__ __forceinline__ void wait_async0() {
  asm volatile("s_wait_asynccnt 0x0" ::: "memory");
}

// problem constants
constexpr int cB = 2, cN = 2048, cDIM = 1024, cH = 16, cD = 64;
constexpr int cMN = cB * cN;                  // 4096 rows

// ---------------------------------------------------------------- fp32 -> bf16
__global__ void cvt_bf16(const float* __restrict__ in, bf16* __restrict__ out, int n) {
  int i = blockIdx.x * 256 + threadIdx.x;
  if (i < n) out[i] = f2bf(in[i]);
}

// ---------------------------------------------------------------- tiled bf16 GEMM
// C[M,N] (fp32) = A[M,K] (bf16, row major) * B[K,N] (bf16, row major) [+ bias]
// block tile 128x128x32, 8 waves, wave tile 64x32; double-buffered async staging
__global__ __launch_bounds__(256) void gemm_bf16(
    const bf16* __restrict__ A, const bf16* __restrict__ Bm,
    float* __restrict__ C, const float* __restrict__ bias,
    int M, int N, int K)
{
  constexpr int TM = 128, TN = 128, TK = 32;
  constexpr int LA = TK + 8;    // 40 bf16 = 80B row stride (16B aligned)
  constexpr int LB = TN + 8;    // 136 bf16 = 272B row stride
  __shared__ __attribute__((aligned(16))) bf16 As[2][TM * LA];
  __shared__ __attribute__((aligned(16))) bf16 Bs[2][TK * LB];

  const int tid  = threadIdx.x;
  const int lane = tid & 31;
  const int wave = tid >> 5;
  const int wm   = wave >> 2;          // 0..1  (64 rows each)
  const int wn   = wave & 3;           // 0..3  (32 cols each)
  const int bm   = blockIdx.y * TM;
  const int bn   = blockIdx.x * TN;
  const int cn   = lane & 15;          // C-layout column
  const int h8   = (lane >> 4) * 8;    // half-wave selector

  // async stage of one 128x32 A tile + 32x128 B tile into buffer `buf`
  auto stage = [&](int buf, int k0) {
#pragma unroll
    for (int c = 0; c < 2; ++c) {
      int ee = tid * 16 + c * 8;
      int ra = ee >> 5, ca = ee & 31;
      async_b128(A + (size_t)(bm + ra) * K + (k0 + ca), &As[buf][ra * LA + ca]);
      int rb = ee >> 7, cb = ee & 127;
      async_b128(Bm + (size_t)(k0 + rb) * N + (bn + cb), &Bs[buf][rb * LB + cb]);
    }
  };

  v8f acc[4][2];
  const v8f vzero = {0.f,0.f,0.f,0.f,0.f,0.f,0.f,0.f};
#pragma unroll
  for (int i = 0; i < 4; ++i)
#pragma unroll
    for (int j = 0; j < 2; ++j) acc[i][j] = vzero;

  stage(0, 0);
  const int nk = K / TK;
  for (int it = 0; it < nk; ++it) {
    const int cur = it & 1;
    wait_async0();          // my 4 async loads for buf[cur] have landed
    __syncthreads();        // everyone's landed; everyone done with buf[cur^1]
    if (it + 1 < nk) stage(cur ^ 1, (it + 1) * TK);  // overlaps with wmma below

    // B fragments: col = lane&15; K rows [h8,h8+8) and [h8+16,h8+24)
    Frag bfr[2];
#pragma unroll
    for (int ni = 0; ni < 2; ++ni) {
#pragma unroll
      for (int s = 0; s < 16; ++s)
        bfr[ni].v[s] = Bs[cur][(h8 + (s & 7) + (s >> 3) * 16) * LB + wn * 32 + ni * 16 + cn];
    }
    // A fragments: row = lane&15; K chunks [h8,h8+8) and [h8+16,h8+24)
#pragma unroll
    for (int mi = 0; mi < 4; ++mi) {
      Frag afr;
      const bf16* p = &As[cur][(wm * 64 + mi * 16 + cn) * LA + h8];
      afr.u[0] = *(const u32x4*)(p);
      afr.u[1] = *(const u32x4*)(p + 16);
#pragma unroll
      for (int ni = 0; ni < 2; ++ni)
        acc[mi][ni] = __builtin_amdgcn_wmma_f32_16x16x32_bf16(
            false, afr.v, false, bfr[ni].v, (short)0, acc[mi][ni], false, false);
    }
  }

  // epilogue: C layout lane l -> M = i + 8*(l>>4), N = l&15
#pragma unroll
  for (int mi = 0; mi < 4; ++mi)
#pragma unroll
    for (int ni = 0; ni < 2; ++ni)
#pragma unroll
      for (int i = 0; i < 8; ++i) {
        int r = bm + wm * 64 + mi * 16 + i + h8;
        int c = bn + wn * 32 + ni * 16 + cn;
        float v = acc[mi][ni][i];
        if (bias) v += bias[c];
        C[(size_t)r * N + c] = v;
      }
}

// ---------------------------------------------------------------- RoPE + repack
// qkv fp32 (B*N, 3*DIM) -> qb/kb/vb bf16 (B,H,N,D); q pre-scaled by 1/sqrt(D)
__global__ void rope_pack(const float* __restrict__ qkv,
                          const float* __restrict__ pos,
                          const float* __restrict__ rf,
                          bf16* __restrict__ qb, bf16* __restrict__ kb,
                          bf16* __restrict__ vb)
{
  int idx = blockIdx.x * 256 + threadIdx.x;       // b*N*H + n*H + h
  if (idx >= cB * cN * cH) return;
  int h = idx % cH;
  int n = (idx / cH) % cN;
  int b = idx / (cH * cN);

  const float* qrow = qkv + (size_t)(b * cN + n) * (3 * cDIM) + h * cD;
  const float* krow = qrow + cDIM;
  const float* vrow = qrow + 2 * cDIM;
  size_t obase = (((size_t)(b * cH + h)) * cN + n) * cD;

  float pp[2] = { pos[(size_t)(b * cN + n) * 2 + 0],
                  pos[(size_t)(b * cN + n) * 2 + 1] };
#pragma unroll
  for (int p = 0; p < 2; ++p) {
#pragma unroll
    for (int f = 0; f < 16; ++f) {
      float th = pp[p] * rf[(p * cH + h) * 16 + f];
      float s, c;
      __sincosf(th, &s, &c);
      int i1 = p * 32 + f, i2 = p * 32 + 16 + f;
      float q1 = qrow[i1], q2 = qrow[i2];
      qb[obase + i1] = f2bf((q1 * c - q2 * s) * 0.125f);   // fold 1/sqrt(64)
      qb[obase + i2] = f2bf((q1 * s + q2 * c) * 0.125f);
      float k1 = krow[i1], k2 = krow[i2];
      kb[obase + i1] = f2bf(k1 * c - k2 * s);
      kb[obase + i2] = f2bf(k1 * s + k2 * c);
    }
  }
#pragma unroll
  for (int d = 0; d < cD; ++d) vb[obase + d] = f2bf(vrow[d]);
}

// ---------------------------------------------------------------- flash attention
// grid (N/128, H, B); 8 waves * 16 queries; 32-key blocks staged in LDS
__global__ __launch_bounds__(256) void attn_flash(
    const bf16* __restrict__ qb, const bf16* __restrict__ kb,
    const bf16* __restrict__ vb, const unsigned char* __restrict__ mask,
    bf16* __restrict__ ob /* (B,N,DIM) */)
{
  constexpr int KB = 32;
  __shared__ __attribute__((aligned(16))) bf16 Kt[cD * KB];        // [d][key]
  __shared__ __attribute__((aligned(16))) bf16 Vs[KB * (cD + 8)];  // [key][d]
  __shared__ __attribute__((aligned(16))) bf16 Ps[8 * 16 * 40];    // per-wave P

  const int tid  = threadIdx.x;
  const int lane = tid & 31;
  const int wave = tid >> 5;
  const int b = blockIdx.z, h = blockIdx.y;
  const int q0 = blockIdx.x * 128 + wave * 16;
  const int cn = lane & 15;
  const int h8 = (lane >> 4) * 8;

  const size_t bh = (size_t)(b * cH + h);
  const bf16* Q  = qb + bh * cN * cD;
  const bf16* Kg = kb + bh * cN * cD;
  const bf16* Vg = vb + bh * cN * cD;
  const unsigned char* mrow = mask + (size_t)b * cN;

  // load the wave's 16x64 Q tile as two 16x32 A-fragments (kept in VGPRs)
  Frag qa[2];
#pragma unroll
  for (int c = 0; c < 2; ++c) {
    const bf16* p = Q + (size_t)(q0 + cn) * cD + c * 32 + h8;
    qa[c].u[0] = *(const u32x4*)(p);
    qa[c].u[1] = *(const u32x4*)(p + 16);
  }

  float m[8], l[8];
  v8f oacc[4];
  const v8f vzero = {0.f,0.f,0.f,0.f,0.f,0.f,0.f,0.f};
#pragma unroll
  for (int i = 0; i < 8; ++i) { m[i] = -1e30f; l[i] = 0.f; }
#pragma unroll
  for (int t = 0; t < 4; ++t) oacc[t] = vzero;

  bf16* Pw = Ps + wave * 16 * 40;

  for (int kb0 = 0; kb0 < cN; kb0 += KB) {
    __syncthreads();
    {   // V tile via async path; K^T tile transposed by hand
      int key2 = tid >> 3, dd = (tid & 7) * 8;
      async_b128(Vg + (size_t)(kb0 + key2) * cD + dd,
                 Vs + key2 * (cD + 8) + dd);
      int key = tid & 31, d0 = (tid >> 5) * 8;
      u32x4 kv = *(const u32x4*)(Kg + (size_t)(kb0 + key) * cD + d0);
      const bf16* kvp = (const bf16*)&kv;
#pragma unroll
      for (int j = 0; j < 8; ++j) Kt[(d0 + j) * KB + key] = kvp[j];
    }
    wait_async0();
    __syncthreads();

    float bias0 = mrow[kb0 + cn]      ? 0.f : -1e30f;
    float bias1 = mrow[kb0 + 16 + cn] ? 0.f : -1e30f;

    // S = Q * K^T : two 16x16 tiles (keys 0..15, 16..31), K-dim 64 = 2 wmma each
    v8f s0 = vzero, s1 = vzero;
#pragma unroll
    for (int c = 0; c < 2; ++c) {
      Frag bk0, bk1;
#pragma unroll
      for (int s = 0; s < 16; ++s) {   // lane&15 = key column, elements over d
        const bf16* p = Kt + (c * 32 + h8 + (s & 7) + (s >> 3) * 16) * KB;
        bk0.v[s] = p[cn];
        bk1.v[s] = p[16 + cn];
      }
      s0 = __builtin_amdgcn_wmma_f32_16x16x32_bf16(false, qa[c].v, false, bk0.v,
                                                   (short)0, s0, false, false);
      s1 = __builtin_amdgcn_wmma_f32_16x16x32_bf16(false, qa[c].v, false, bk1.v,
                                                   (short)0, s1, false, false);
    }

    // online softmax; rows live per-VGPR across 16-lane halves
    float alpha[8];
#pragma unroll
    for (int i = 0; i < 8; ++i) {
      float v0 = s0[i] + bias0, v1 = s1[i] + bias1;
      float bm_ = fmaxf(v0, v1);
#pragma unroll
      for (int off = 1; off < 16; off <<= 1)
        bm_ = fmaxf(bm_, __shfl_xor(bm_, off, 32));
      float mn = fmaxf(m[i], bm_);
      float a_ = __expf(m[i] - mn);
      float p0 = __expf(v0 - mn);
      float p1 = __expf(v1 - mn);
      float rs = p0 + p1;
#pragma unroll
      for (int off = 1; off < 16; off <<= 1) rs += __shfl_xor(rs, off, 32);
      l[i] = l[i] * a_ + rs;
      m[i] = mn;
      alpha[i] = a_;
      int r = i + h8;                       // C-layout row
      Pw[r * 40 + cn]      = f2bf(p0);
      Pw[r * 40 + 16 + cn] = f2bf(p1);
    }
#pragma unroll
    for (int t = 0; t < 4; ++t)
#pragma unroll
      for (int i = 0; i < 8; ++i) oacc[t][i] = oacc[t][i] * alpha[i];

    // reload P (16x32) as A-fragment; same-wave LDS store->load fence
    asm volatile("s_wait_dscnt 0x0" ::: "memory");
    Frag pa;
    pa.u[0] = *(const u32x4*)(Pw + cn * 40 + h8);
    pa.u[1] = *(const u32x4*)(Pw + cn * 40 + h8 + 16);

    // O += P * V : four 16x16 tiles over d
#pragma unroll
    for (int t = 0; t < 4; ++t) {
      Frag bv;
#pragma unroll
      for (int s = 0; s < 16; ++s)
        bv.v[s] = Vs[(h8 + (s & 7) + (s >> 3) * 16) * (cD + 8) + t * 16 + cn];
      oacc[t] = __builtin_amdgcn_wmma_f32_16x16x32_bf16(false, pa.v, false, bv.v,
                                                        (short)0, oacc[t], false, false);
    }
  }

  // normalize and write to (B,N,DIM) bf16 for the output projection
#pragma unroll
  for (int t = 0; t < 4; ++t)
#pragma unroll
    for (int i = 0; i < 8; ++i) {
      int r = q0 + i + h8;
      int c = h * cD + t * 16 + cn;
      float inv = 1.0f / l[i];
      ob[((size_t)b * cN + r) * cDIM + c] = f2bf(oacc[t][i] * inv);
    }
}

// ---------------------------------------------------------------- launcher
extern "C" void kernel_launch(void* const* d_in, const int* in_sizes, int n_in,
                              void* d_out, int out_size, void* d_ws, size_t ws_size,
                              hipStream_t stream) {
  const float* x     = (const float*)d_in[0];
  const float* w_qkv = (const float*)d_in[1];
  const float* w_out = (const float*)d_in[2];
  const float* b_out = (const float*)d_in[3];
  const float* rf    = (const float*)d_in[4];
  const float* pos   = (const float*)d_in[5];
  const unsigned char* mask = (const unsigned char*)d_in[6];
  float* out = (float*)d_out;

  char* w = (char*)d_ws;
  bf16* xb    = (bf16*)w;  w += (size_t)cMN * cDIM * 2;
  bf16* wqkvb = (bf16*)w;  w += (size_t)cDIM * 3 * cDIM * 2;
  bf16* woutb = (bf16*)w;  w += (size_t)cDIM * cDIM * 2;
  float* qkv  = (float*)w; w += (size_t)cMN * 3 * cDIM * 4;
  bf16* qb    = (bf16*)w;  w += (size_t)cMN * cDIM * 2;
  bf16* kb    = (bf16*)w;  w += (size_t)cMN * cDIM * 2;
  bf16* vb    = (bf16*)w;  w += (size_t)cMN * cDIM * 2;
  bf16* ob    = (bf16*)w;  w += (size_t)cMN * cDIM * 2;

  int nx = cMN * cDIM;           // 4,194,304
  int nw = cDIM * 3 * cDIM;      // 3,145,728
  int no = cDIM * cDIM;          // 1,048,576
  cvt_bf16<<<(nx + 255) / 256, 256, 0, stream>>>(x, xb, nx);
  cvt_bf16<<<(nw + 255) / 256, 256, 0, stream>>>(w_qkv, wqkvb, nw);
  cvt_bf16<<<(no + 255) / 256, 256, 0, stream>>>(w_out, woutb, no);

  // qkv = x @ w_qkv  (4096 x 3072, K=1024)
  gemm_bf16<<<dim3(3 * cDIM / 128, cMN / 128), 256, 0, stream>>>(
      xb, wqkvb, qkv, nullptr, cMN, 3 * cDIM, cDIM);

  // RoPE + repack to (B,H,N,D)
  rope_pack<<<(cB * cN * cH + 255) / 256, 256, 0, stream>>>(qkv, pos, rf, qb, kb, vb);

  // flash attention
  attn_flash<<<dim3(cN / 128, cH, cB), 256, 0, stream>>>(qb, kb, vb, mask, ob);

  // out = o @ w_out + b_out  (4096 x 1024, K=1024)
  gemm_bf16<<<dim3(cDIM / 128, cMN / 128), 256, 0, stream>>>(
      ob, woutb, out, b_out, cMN, cDIM, cDIM);
}
